// ActivationGATSingleHeadLayer_83476984365544
// MI455X (gfx1250) — matrix operand order repaired and checked by reading, compiled, not run on hardware
//
#include <hip/hip_runtime.h>
#include <hip/hip_bf16.h>
#include <math.h>

typedef __attribute__((ext_vector_type(16))) _Float16 v16h;
typedef __attribute__((ext_vector_type(8)))  float    v8f;

#define D_FEAT 64
#define BN_EPS 1e-5f

// ---------------------------------------------------------------------------
// K0: zero-fill helper (d_out accumulator + ws control buffers)
// ---------------------------------------------------------------------------
__global__ void zero_f32(float* __restrict__ p, long n) {
  long i = (long)blockIdx.x * blockDim.x + threadIdx.x;
  const long stride = (long)gridDim.x * blockDim.x;
  for (; i < n; i += stride) p[i] = 0.0f;
}

// ---------------------------------------------------------------------------
// K1: per-edge dot products via WMMA. One wave handles a tile of 16 edges.
//   A (16x32 f16) row m = z[src[tile*16+m]] (K-chunk), B (32x16 f16) col n =
//   z[dst[tile*16+n]]. D = A*B accumulated fp32 over K=64; diag(D) = edge dots.
//   e = relu(dot); segment-max into emax via uint atomicMax (valid: e >= 0).
// ---------------------------------------------------------------------------
__global__ __launch_bounds__(256)
void edge_dot_wmma(const float* __restrict__ z,
                   const int* __restrict__ src,
                   const int* __restrict__ dst,
                   float* __restrict__ e,
                   unsigned int* __restrict__ emax_bits,
                   int E) {
  const int lane = threadIdx.x & 31;
  const int wave = threadIdx.x >> 5;
  const long tile = (long)blockIdx.x * (blockDim.x >> 5) + wave;
  const long base = tile * 16;
  if (base >= E) return;  // wave-uniform exit: EXEC stays all-ones for WMMA

  // Lane r supplies A-row r (from z[src]) and B-column r (from z[dst]).
  const int  r     = lane & 15;
  const long eidx  = min(base + (long)r, (long)E - 1);
  const float* rowS = z + (size_t)src[eidx] * D_FEAT;
  const float* rowD = z + (size_t)dst[eidx] * D_FEAT;

  // ISA per-lane K offsets (16-bit A 16x32 and B 32x16 layouts):
  const int aOff = (lane < 16) ? 0 : 8;   // A: elems 0..7 -> K=aOff+j, 8..15 -> K=aOff+16+j
  const int bOff = (lane < 16) ? 0 : 8;   // B: elems 0..7 -> K=bOff+j, 8..15 -> K=bOff+16+j

  v8f c = {};
#pragma unroll
  for (int kb = 0; kb < D_FEAT; kb += 32) {
    float4 A0 = *(const float4*)(rowS + kb + aOff + 0);
    float4 A1 = *(const float4*)(rowS + kb + aOff + 4);
    float4 A2 = *(const float4*)(rowS + kb + aOff + 16);
    float4 A3 = *(const float4*)(rowS + kb + aOff + 20);
    float4 B0 = *(const float4*)(rowD + kb + bOff + 0);
    float4 B1 = *(const float4*)(rowD + kb + bOff + 4);
    float4 B2 = *(const float4*)(rowD + kb + bOff + 16);
    float4 B3 = *(const float4*)(rowD + kb + bOff + 20);

    v16h a, b;
    a[0]=(_Float16)A0.x; a[1]=(_Float16)A0.y; a[2]=(_Float16)A0.z; a[3]=(_Float16)A0.w;
    a[4]=(_Float16)A1.x; a[5]=(_Float16)A1.y; a[6]=(_Float16)A1.z; a[7]=(_Float16)A1.w;
    a[8]=(_Float16)A2.x; a[9]=(_Float16)A2.y; a[10]=(_Float16)A2.z; a[11]=(_Float16)A2.w;
    a[12]=(_Float16)A3.x; a[13]=(_Float16)A3.y; a[14]=(_Float16)A3.z; a[15]=(_Float16)A3.w;
    b[0]=(_Float16)B0.x; b[1]=(_Float16)B0.y; b[2]=(_Float16)B0.z; b[3]=(_Float16)B0.w;
    b[4]=(_Float16)B1.x; b[5]=(_Float16)B1.y; b[6]=(_Float16)B1.z; b[7]=(_Float16)B1.w;
    b[8]=(_Float16)B2.x; b[9]=(_Float16)B2.y; b[10]=(_Float16)B2.z; b[11]=(_Float16)B2.w;
    b[12]=(_Float16)B3.x; b[13]=(_Float16)B3.y; b[14]=(_Float16)B3.z; b[15]=(_Float16)B3.w;

    // D(f32 16x16) = A(f16) x B(f16) + C
    c = __builtin_amdgcn_wmma_f32_16x16x32_f16(
        /*neg_a=*/false, a, /*neg_b=*/false, b,
        /*c_mod=*/(short)0, c, /*reuse_a=*/false, /*reuse_b=*/false);
  }

  // Diagonal extraction per the 32-bit 16x16 C/D layout:
  //   M=N=m<8  -> lane m,    VGPR m
  //   M=N=m>=8 -> lane m+16, VGPR m-8
  int m = -1, vg = 0;
  if (lane < 8)        { m = lane;      vg = lane;      }
  else if (lane >= 24) { m = lane - 16; vg = lane - 24; }
  if (m >= 0) {
    const long ei = base + m;
    if (ei < E) {
      float val = 0.0f;
#pragma unroll
      for (int q = 0; q < 8; ++q) val = (vg == q) ? c[q] : val;
      val = fmaxf(val, 0.0f);  // relu
      e[ei] = val;
      atomicMax(&emax_bits[dst[ei]], __float_as_uint(val));
    }
  }
}

// ---------------------------------------------------------------------------
// K2: e_exp = exp(e - emax[dst]); denom[dst] += e_exp
// ---------------------------------------------------------------------------
__global__ void edge_exp_denom(const float* __restrict__ e,
                               const int* __restrict__ dst,
                               const float* __restrict__ emax,
                               float* __restrict__ e_exp,
                               float* __restrict__ denom,
                               int E) {
  const long i = (long)blockIdx.x * blockDim.x + threadIdx.x;
  if (i >= E) return;
  const int d = dst[i];
  const float ee = expf(e[i] - emax[d]);  // arg <= 0
  e_exp[i] = ee;
  atomicAdd(&denom[d], ee);
}

// ---------------------------------------------------------------------------
// K3: h[dst] += (e_exp/denom[dst]) * z[src]; 64 threads per edge so the row
// read and the atomic burst are each 256B contiguous.
// ---------------------------------------------------------------------------
__global__ __launch_bounds__(256)
void aggregate(const float* __restrict__ z,
               const int* __restrict__ src,
               const int* __restrict__ dst,
               const float* __restrict__ e_exp,
               const float* __restrict__ denom,
               float* __restrict__ h,
               int E) {
  const int t = threadIdx.x;
  const long edge = (long)blockIdx.x * (blockDim.x >> 6) + (t >> 6);
  const int f = t & 63;
  if (edge >= E) return;
  const int d = dst[edge];
  const float alpha = e_exp[edge] / denom[d];
  const float v = alpha * z[(size_t)src[edge] * D_FEAT + f];
  atomicAdd(&h[(size_t)d * D_FEAT + f], v);
}

// ---------------------------------------------------------------------------
// K4: per-feature sum / sum-of-squares over nodes (LDS-combine, then 128
// global atomics per block).
// ---------------------------------------------------------------------------
__global__ __launch_bounds__(256)
void bn_stats(const float* __restrict__ h, float* __restrict__ stats, int N) {
  __shared__ float s_sum[D_FEAT];
  __shared__ float s_sq[D_FEAT];
  const int t = threadIdx.x;
  if (t < D_FEAT) { s_sum[t] = 0.0f; s_sq[t] = 0.0f; }
  __syncthreads();

  const int f = t & 63;
  const int rowsPerBlk = blockDim.x >> 6;  // 4
  float sum = 0.0f, sq = 0.0f;
  for (long r = (long)blockIdx.x * rowsPerBlk + (t >> 6); r < N;
       r += (long)gridDim.x * rowsPerBlk) {
    const float v = h[r * D_FEAT + f];
    sum += v;
    sq  += v * v;
  }
  atomicAdd(&s_sum[f], sum);
  atomicAdd(&s_sq[f], sq);
  __syncthreads();
  if (t < D_FEAT) {
    atomicAdd(&stats[t], s_sum[t]);
    atomicAdd(&stats[D_FEAT + t], s_sq[t]);
  }
}

// ---------------------------------------------------------------------------
// K5: out = relu((h - mean) * rsqrt(var + eps) * gamma + beta)  (in place)
// ---------------------------------------------------------------------------
__global__ void bn_relu(float* __restrict__ h,
                        const float* __restrict__ stats,
                        const float* __restrict__ gamma,
                        const float* __restrict__ beta,
                        int N) {
  const long i = (long)blockIdx.x * blockDim.x + threadIdx.x;
  if (i >= (long)N * D_FEAT) return;
  const int f = (int)(i & 63);
  const float invN = 1.0f / (float)N;
  const float mean = stats[f] * invN;
  const float var = stats[D_FEAT + f] * invN - mean * mean;  // biased var
  const float inv = rsqrtf(var + BN_EPS);
  const float v = (h[i] - mean) * inv * gamma[f] + beta[f];
  h[i] = fmaxf(v, 0.0f);
}

// ---------------------------------------------------------------------------
extern "C" void kernel_launch(void* const* d_in, const int* in_sizes, int n_in,
                              void* d_out, int out_size, void* d_ws, size_t ws_size,
                              hipStream_t stream) {
  const float* z     = (const float*)d_in[0];
  const int*   src   = (const int*)d_in[1];
  const int*   dst   = (const int*)d_in[2];
  const float* gamma = (const float*)d_in[3];
  const float* beta  = (const float*)d_in[4];

  const int N = in_sizes[0] / D_FEAT;
  const int E = in_sizes[1];

  float* h = (float*)d_out;  // accumulate h directly in the output buffer

  // Workspace layout (floats): e[E] | e_exp[E] | emax[N] | denom[N] | stats[128]
  float* ws      = (float*)d_ws;
  float* e       = ws;
  float* e_exp   = ws + (size_t)E;
  float* emax    = ws + (size_t)2 * E;
  float* denom   = emax + N;
  float* stats   = denom + N;

  // K0: zero output accumulator and control buffers (emax/denom/stats are
  // contiguous: 2N + 128 floats).
  zero_f32<<<1024, 256, 0, stream>>>(h, (long)N * D_FEAT);
  zero_f32<<<256, 256, 0, stream>>>(emax, (long)2 * N + 2 * D_FEAT);

  // K1: WMMA edge dots + relu + segment-max
  {
    const long tiles = ((long)E + 15) / 16;
    const int wavesPerBlock = 8;  // 256 threads
    const int blocks = (int)((tiles + wavesPerBlock - 1) / wavesPerBlock);
    edge_dot_wmma<<<blocks, 256, 0, stream>>>(z, src, dst, e,
                                              (unsigned int*)emax, E);
  }

  // K2: softmax numerators + denominators
  edge_exp_denom<<<(E + 255) / 256, 256, 0, stream>>>(e, dst, emax, e_exp,
                                                      denom, E);

  // K3: weighted scatter-add aggregation into h
  {
    const int edgesPerBlock = 4;  // 256 threads / 64 feat
    const int blocks = (E + edgesPerBlock - 1) / edgesPerBlock;
    aggregate<<<blocks, 256, 0, stream>>>(z, src, dst, e_exp, denom, h, E);
  }

  // K4 + K5: BatchNorm (training-mode batch stats) + relu, in place on d_out
  bn_stats<<<256, 256, 0, stream>>>(h, stats, N);
  bn_relu<<<((long)N * D_FEAT + 255) / 256, 256, 0, stream>>>(h, stats, gamma,
                                                              beta, N);
}
